// PatchMerging3D_43490838840019
// MI455X (gfx1250) — matrix operand order repaired and checked by reading, not compiled, not run on hardware
//
#include <hip/hip_runtime.h>
#include <stdint.h>

typedef __attribute__((ext_vector_type(16))) _Float16 v16h;
typedef __attribute__((ext_vector_type(8)))  float    v8f;
typedef __attribute__((ext_vector_type(4)))  unsigned v4u;
typedef __attribute__((ext_vector_type(8)))  int      v8i;
typedef __attribute__((ext_vector_type(4)))  int      v4i;

#define CCH   384          // C
#define K8C   3072         // 8*C (GEMM K)
#define OC    768          // 2*C (GEMM N)
#define NROWS 16384        // 4*16*16*16 (GEMM M)
#define LDAH  40           // halves per LDS row: 32 data + 4-DWORD TDM pad -> 80B stride
#define TILE_HALVES (128 * LDAH)
#define TILE_BYTES  (TILE_HALVES * 2)   // 10240 B per buffer

// ---------------- Kernel 1: patch-merge + LayerNorm + f16 convert ----------------
__global__ __launch_bounds__(256) void pm_ln_f16_kernel(
    const float* __restrict__ x, const float* __restrict__ gamma,
    const float* __restrict__ beta, _Float16* __restrict__ a16)
{
  const int r = blockIdx.x;            // merged row 0..16383
  const int t = threadIdx.x;           // 0..255
  const int b  = r >> 12;
  const int h2 = (r >> 8) & 15;
  const int w2 = (r >> 4) & 15;
  const int d2 = r & 15;

  float v[12];
  float sum = 0.f, sumsq = 0.f;
#pragma unroll
  for (int e = 0; e < 12; ++e) {
    const int idx   = t + (e << 8);            // 0..3071
    const int chunk = idx / CCH;               // 0..7 = i*4 + j*2 + k
    const int c     = idx - chunk * CCH;
    const int hh = 2*h2 + ((chunk >> 2) & 1);
    const int ww = 2*w2 + ((chunk >> 1) & 1);
    const int dd = 2*d2 + ( chunk       & 1);
    const float val = x[(((size_t)((b*32 + hh)*32 + ww))*32 + dd)*CCH + c];
    v[e] = val;
    sum += val; sumsq += val*val;
  }

  __shared__ float s1[256], s2[256];
  s1[t] = sum; s2[t] = sumsq;
  __syncthreads();
  for (int off = 128; off > 0; off >>= 1) {
    if (t < off) { s1[t] += s1[t+off]; s2[t] += s2[t+off]; }
    __syncthreads();
  }
  const float mu   = s1[0] * (1.0f / K8C);
  const float var  = s2[0] * (1.0f / K8C) - mu * mu;
  const float rstd = rsqrtf(var + 1e-5f);

#pragma unroll
  for (int e = 0; e < 12; ++e) {
    const int idx = t + (e << 8);
    const float xn = (v[e] - mu) * rstd * gamma[idx] + beta[idx];
    a16[(size_t)r * K8C + idx] = (_Float16)xn;
  }
}

// ---------------- Kernel 2: weight fp32 -> f16 ----------------
__global__ __launch_bounds__(256) void wcvt_kernel(
    const float* __restrict__ w, _Float16* __restrict__ wh)
{
  const int i = (blockIdx.x * 256 + threadIdx.x) * 4;   // OC*K8C divisible by 1024
  const float4 f = *(const float4*)(w + i);
  wh[i+0] = (_Float16)f.x;
  wh[i+1] = (_Float16)f.y;
  wh[i+2] = (_Float16)f.z;
  wh[i+3] = (_Float16)f.w;
}

// ---------------- TDM 2-D tile load: one descriptor per 128x32-half tile ----------------
// D# per CDNA5 ISA; amdgpu-toolchain 6-arg builtin
// (v4u g0, v8i g1, v4i g2, v4i g3, v8i, i32 cpol).
__device__ __forceinline__ void tdm_load_2d(unsigned lds_addr, const _Float16* gaddr,
                                            unsigned tensor_rows)
{
  const uint64_t ga = (uint64_t)(uintptr_t)gaddr;
  v4u g0;
  g0[0] = 1u;                                              // count=1 (user mode)
  g0[1] = lds_addr;                                        // lds_addr[31:0]
  g0[2] = (unsigned)ga;                                    // global_addr[31:0]
  g0[3] = (unsigned)((ga >> 32) & 0x01FFFFFFu) | (2u << 30); // addr[56:32] | type=2

  v8i g1;
  // workgroup_mask=0 (not in cluster); data_size=1 (2B); pad_enable;
  // pad_interval=3 (16 DWORDs = one 64B row); pad_amount=3 (4 DWORDs) -> 80B row stride
  g1[0] = (int)((1u << 16) | (1u << 20) | (3u << 22) | (3u << 25));
  g1[1] = (int)(((unsigned)K8C & 0xFFFFu) << 16);          // tensor_dim0[15:0] (atomic_addr=0)
  g1[2] = (int)((((unsigned)K8C >> 16) & 0xFFFFu) | ((tensor_rows & 0xFFFFu) << 16));
  g1[3] = (int)(((tensor_rows >> 16) & 0xFFFFu) | (32u << 16));  // tile_dim0 = 32 halves
  g1[4] = (int)(128u);                                     // tile_dim1 = 128 rows; tile_dim2=0
  g1[5] = (int)K8C;                                        // tensor_dim0_stride[31:0]
  g1[6] = 0;                                               // stride[47:32]=0; dim1_stride lo=0
  g1[7] = 0;

  const v4i z4 = {0, 0, 0, 0};                             // groups 2/3 unused (2-D tensor)
  const v8i z8 = {0, 0, 0, 0, 0, 0, 0, 0};
  __builtin_amdgcn_tensor_load_to_lds(g0, g1, z4, z4, z8, 0);
}

// ---------------- Kernel 3: f16 WMMA GEMM, TDM-fed, triple-buffered LDS ----------------
// Y[M,N] = A[M,K] * W[N,K]^T ; block tile 128x128, 8 waves of 64x32 each.
// K loop unrolled by 3 so buffer rotation is compile-time (no %3 math in wave 0's path).
__global__ __launch_bounds__(256) void gemm_wmma_tdm_kernel(
    const _Float16* __restrict__ A, const _Float16* __restrict__ W,
    float* __restrict__ Y)
{
  __shared__ __align__(16) _Float16 As[3 * TILE_HALVES];
  __shared__ __align__(16) _Float16 Bs[3 * TILE_HALVES];

  const int t    = threadIdx.x;
  const int lane = t & 31;
  const int wave = t >> 5;
  const int lo   = lane & 15;
  const int hi   = lane >> 4;
  const int wm   = wave >> 2;          // 0..1 : M offset wm*64
  const int wn   = wave & 3;           // 0..3 : N offset wn*32

  const int n0 = blockIdx.x * 128;     // N fastest: W stays L2-resident, A tile reused
  const int m0 = blockIdx.y * 128;

  const unsigned ldsA = (unsigned)(uintptr_t)&As[0];   // flat->LDS: low 32 bits are LDS offset
  const unsigned ldsB = (unsigned)(uintptr_t)&Bs[0];

  const _Float16* __restrict__ Abase = A + (size_t)m0 * K8C;
  const _Float16* __restrict__ Bbase = W + (size_t)n0 * K8C;

  // LDS operand addresses are loop-invariant per buffer; precompute base offsets (halves).
  const int aRow = (wm * 64 + lo) * LDAH + 8 * hi;         // + mi*16*LDAH, +0/+16
  const int bCol = (wn * 32 + lo) * LDAH + 8 * hi;         // + ni*16*LDAH, +0/+16

  v8f acc[4][2] = {};

  if (wave == 0) {                     // TDM ignores EXEC; issued via wave-level branch
    tdm_load_2d(ldsA, Abase, NROWS);
    tdm_load_2d(ldsB, Bbase, OC);
  }

  auto step = [&](int i, int cb, int nb) {
    const int kn = (i + 1) * 32;
    if (wave == 0) {
      if (kn < K8C) {
        tdm_load_2d(ldsA + (unsigned)(nb * TILE_BYTES), Abase + kn, NROWS);
        tdm_load_2d(ldsB + (unsigned)(nb * TILE_BYTES), Bbase + kn, OC);
        __builtin_amdgcn_s_wait_tensorcnt(2);   // only newest pair may stay in flight
      } else {
        __builtin_amdgcn_s_wait_tensorcnt(0);
      }
    }
    __syncthreads();   // publishes buf cb; fences reads of buf nb (last read 2 iters ago)

    const _Float16* Ab = As + cb * TILE_HALVES;
    const _Float16* Bb = Bs + cb * TILE_HALVES;

    // CDNA5 dense 16-bit A (16x32): lane-hi picks K halves {8h..8h+7, 16+8h..16+8h+7}
    // CDNA5 dense 16-bit B (32x16): lane-hi picks K halves {8h..8h+7, 16+8h..16+8h+7}
    union { v16h h; float4 f[2]; } af[4], bf[2];
#pragma unroll
    for (int mi = 0; mi < 4; ++mi) {
      af[mi].f[0] = *(const float4*)(Ab + aRow + mi * 16 * LDAH);
      af[mi].f[1] = *(const float4*)(Ab + aRow + mi * 16 * LDAH + 16);
    }
#pragma unroll
    for (int ni = 0; ni < 2; ++ni) {
      bf[ni].f[0] = *(const float4*)(Bb + bCol + ni * 16 * LDAH);
      bf[ni].f[1] = *(const float4*)(Bb + bCol + ni * 16 * LDAH + 16);
    }
#pragma unroll
    for (int mi = 0; mi < 4; ++mi)
#pragma unroll
      for (int ni = 0; ni < 2; ++ni)
        acc[mi][ni] = __builtin_amdgcn_wmma_f32_16x16x32_f16(
            false, af[mi].h, false, bf[ni].h, (short)0, acc[mi][ni], false, false);
  };

  for (int i = 0; i < K8C / 32; i += 3) {   // 96 steps, triple-buffer rotation 0->1->2
    step(i + 0, 0, 1);
    step(i + 1, 1, 2);
    step(i + 2, 2, 0);
  }

  // C/D layout: VGPR r, lanes 0-15 -> M=r, lanes 16-31 -> M=r+8, N=lane&15
  const int rbase = m0 + wm * 64 + (hi ? 8 : 0);
#pragma unroll
  for (int mi = 0; mi < 4; ++mi) {
#pragma unroll
    for (int ni = 0; ni < 2; ++ni) {
      const int col = n0 + wn * 32 + ni * 16 + lo;
#pragma unroll
      for (int r = 0; r < 8; ++r)
        Y[(size_t)(rbase + mi * 16 + r) * OC + col] = acc[mi][ni][r];
    }
  }
}

extern "C" void kernel_launch(void* const* d_in, const int* in_sizes, int n_in,
                              void* d_out, int out_size, void* d_ws, size_t ws_size,
                              hipStream_t stream) {
  const float* x     = (const float*)d_in[0];
  const float* gamma = (const float*)d_in[1];
  const float* beta  = (const float*)d_in[2];
  const float* w     = (const float*)d_in[3];
  float* y = (float*)d_out;

  _Float16* a16 = (_Float16*)d_ws;                       // 16384*3072*2B = 100.7 MB
  _Float16* w16 = a16 + (size_t)NROWS * K8C;             // + 768*3072*2B = 4.7 MB

  pm_ln_f16_kernel<<<NROWS, 256, 0, stream>>>(x, gamma, beta, a16);
  wcvt_kernel<<<(OC * K8C) / 1024, 256, 0, stream>>>(w, w16);
  gemm_wmma_tdm_kernel<<<dim3(OC / 128, NROWS / 128), 256, 0, stream>>>(a16, w16, y);
}
